// EfficientAttention_21028159881811
// MI455X (gfx1250) — matrix operand order repaired and checked by reading, not compiled, not run on hardware
//
#include <hip/hip_runtime.h>
#include <hip/hip_bf16.h>

// ---------------------------------------------------------------------------
// EfficientAttention (linear attention) for MI455X / gfx1250, wave32 + WMMA.
// N=8 batches, L=HW=16384, D=256 channels, H=8 heads, hk=hv=32.
// GEMMs on v_wmma_f32_16x16x32_bf16; context stage uses the Tensor Data Mover
// (tensor_load_to_lds + s_wait_tensorcnt) with LDS row padding.
// ---------------------------------------------------------------------------

#define NB 8
#define DD 256
#define LL 16384
#define HH 8

typedef __attribute__((ext_vector_type(16))) __bf16 v16bf;
typedef __attribute__((ext_vector_type(8)))  float  v8f;
typedef __attribute__((ext_vector_type(4)))  unsigned int u32x4;
typedef __attribute__((ext_vector_type(8)))  int i32x8;
typedef __attribute__((ext_vector_type(4)))  int i32x4;

union FragBF {
    v16bf v;
    uint4 q[2];
    unsigned short s[16];
};

__device__ __forceinline__ unsigned short f32_to_bf16(float f) {
    unsigned int u = __float_as_uint(f);
    unsigned int r = u + 0x7FFFu + ((u >> 16) & 1u); // round to nearest even
    return (unsigned short)(r >> 16);
}

__device__ __forceinline__ v8f wmma_bf16(const FragBF& a, const FragBF& b, v8f c) {
    return __builtin_amdgcn_wmma_f32_16x16x32_bf16(
        false, a.v, false, b.v, (short)0, c, false, false);
}

// ---- TDM: 2D tile load global->LDS (bf16 elements), D# per ISA 08 §8.3/8.4
// w0_flags carries {data_size, pad_enable, pad_interval, pad_amount} bits.
// This toolchain exposes the 6-arg builtin: (g0, g1, g2, g3, g4, cpol).
__device__ __forceinline__ void tdm_load_2d_bf16(
    unsigned lds_off, const unsigned short* gptr,
    unsigned tensor_d0, unsigned tensor_d1,
    unsigned tile_d0, unsigned tile_d1,
    unsigned long long stride_elems, unsigned w0_flags) {
    unsigned long long ga = (unsigned long long)(size_t)gptr;
    u32x4 g0;
    g0.x = 1u;                                     // count=1, user descriptor
    g0.y = lds_off;                                // D#.lds_addr (bytes)
    g0.z = (unsigned)ga;                           // global_addr[31:0]
    g0.w = (unsigned)((ga >> 32) & 0x01FFFFFFu) | 0x80000000u; // addr[56:32]|type=2
    i32x8 g1;
    g1[0] = (int)w0_flags;                         // wg_mask=0 | data_size | pad cfg
    g1[1] = (int)((tensor_d0 & 0xFFFFu) << 16);    // tensor_dim0[15:0] @bit48
    g1[2] = (int)((tensor_d0 >> 16) | ((tensor_d1 & 0xFFFFu) << 16));
    g1[3] = (int)((tensor_d1 >> 16) | (tile_d0 << 16)); // tile_dim0 @bit112
    g1[4] = (int)(tile_d1 & 0xFFFFu);              // tile_dim1 (tile_dim2=0)
    g1[5] = (int)(stride_elems & 0xFFFFFFFFu);     // tensor_dim0_stride[31:0]
    g1[6] = (int)((stride_elems >> 32) & 0xFFFFu); // stride[47:32]
    g1[7] = 0;                                     // tensor_dim1_stride unused
    i32x4 z4 = {0, 0, 0, 0};
    i32x8 z8 = {0, 0, 0, 0, 0, 0, 0, 0};
    __builtin_amdgcn_tensor_load_to_lds(g0, g1, z4, z4, z8, 0);
}

// data_size=1(2B)<<16 | pad_enable<<20 | pad_interval=6(128DW)<<22 | pad_amount=3(4DW)<<25
#define TDM_FLAGS_256COL 0x07910000u
#define CTX_PITCH 264   // 256 data + 4 DW pad = 264 u16 = 528B (16B aligned)

// --------------------------- f32 -> bf16 convert ---------------------------
__global__ __launch_bounds__(256) void convert_bf16_kernel(
    const float* __restrict__ in, unsigned short* __restrict__ out, unsigned int n) {
    unsigned int i = blockIdx.x * 256u + threadIdx.x;
    if (i < n) out[i] = f32_to_bf16(in[i]);
}

// ------------------- big GEMM: C[m,l] = sum_d W[m,d]*X[d,l] + b[m] ---------
// Block = 256 thr = 8 waves; tile M=16 x N=512, wave tile 16x64.
// X tile (32k x 512l) staged TRANSPOSED into LDS T[l][k], pitch 40 u16 (80B).
// Transpose store packs 4 k per ds_store_b64.
__global__ __launch_bounds__(256) void gemm_proj_kernel(
    const unsigned short* __restrict__ W, const float* __restrict__ bias,
    const unsigned short* __restrict__ X,
    float* __restrict__ outF, unsigned short* __restrict__ outB) {
    __shared__ __align__(16) unsigned short T[512 * 40]; // 40 KB

    const int l0   = blockIdx.x * 512;
    const int m0   = blockIdx.y * 16;
    const int nb   = blockIdx.z;
    const int t    = threadIdx.x;
    const int wave = t >> 5;
    const int lane = t & 31;
    const int ml   = lane & 15;   // M for A, N for B, col for C/D
    const int half = lane >> 4;   // K-half selector

    const unsigned short* Xb = X + (size_t)nb * DD * LL;

    v8f acc[4];
    for (int c = 0; c < 4; ++c) acc[c] = (v8f){0,0,0,0,0,0,0,0};

    const int kg   = t & 7;   // k-group: rows 4*kg .. 4*kg+3
    const int lrow = t >> 3;  // 0..31 : columns lrow*16 .. +15

    for (int k0 = 0; k0 < DD; k0 += 32) {
        // ---- load 4 rows x 16 cols, transpose-pack into b64 LDS stores
        union { uint4 q[2]; unsigned short s[16]; } rb[4];
        const unsigned short* src = Xb + (size_t)(k0 + kg * 4) * LL + l0 + lrow * 16;
        #pragma unroll
        for (int r = 0; r < 4; ++r) {
            rb[r].q[0] = ((const uint4*)(src + (size_t)r * LL))[0];
            rb[r].q[1] = ((const uint4*)(src + (size_t)r * LL))[1];
        }
        if (k0 + 32 < DD)
            __builtin_prefetch(src + (size_t)32 * LL, 0, 1);
        __syncthreads();               // previous iteration's reads done
        #pragma unroll
        for (int j = 0; j < 16; ++j) {
            ushort4 p;
            p.x = rb[0].s[j]; p.y = rb[1].s[j]; p.z = rb[2].s[j]; p.w = rb[3].s[j];
            *(ushort4*)&T[(lrow * 16 + j) * 40 + kg * 4] = p;
        }
        __syncthreads();               // tile visible to all waves

        // ---- A fragment: W rows m0..m0+15, K = k0..k0+31
        FragBF a;
        const unsigned short* wr = W + (size_t)(m0 + ml) * DD + k0;
        a.q[0] = *(const uint4*)(wr + 16 * half);
        a.q[1] = *(const uint4*)(wr + 16 * half + 8);

        // ---- 4 WMMAs covering this wave's 64 columns
        const int nbase = wave * 64;
        #pragma unroll
        for (int c = 0; c < 4; ++c) {
            FragBF b;
            const unsigned short* trow = &T[(nbase + c * 16 + ml) * 40];
            b.q[0] = *(const uint4*)(trow + 16 * half);
            b.q[1] = *(const uint4*)(trow + 16 * half + 8);
            acc[c] = wmma_bf16(a, b, acc[c]);
        }
    }

    // ---- epilogue: bias + store (branches hoisted, uniform)
    const size_t base = (size_t)nb * DD * LL;
    if (outF) {
        #pragma unroll
        for (int c = 0; c < 4; ++c) {
            const int col = l0 + wave * 64 + c * 16 + ml;
            #pragma unroll
            for (int r = 0; r < 8; ++r) {
                const int row = m0 + r + 8 * half;
                outF[base + (size_t)row * LL + col] = acc[c][r] + bias[row];
            }
        }
    }
    if (outB) {
        #pragma unroll
        for (int c = 0; c < 4; ++c) {
            const int col = l0 + wave * 64 + c * 16 + ml;
            #pragma unroll
            for (int r = 0; r < 8; ++r) {
                const int row = m0 + r + 8 * half;
                outB[base + (size_t)row * LL + col] = f32_to_bf16(acc[c][r] + bias[row]);
            }
        }
    }
}

// ----------------- softmax over L (16384) per (n,channel) row --------------
__global__ __launch_bounds__(256) void softmax_rows_kernel(
    const float* __restrict__ in, unsigned short* __restrict__ outbf) {
    __shared__ float red[256];
    const size_t row = blockIdx.x;          // 0 .. NB*DD-1
    const float* p = in + row * LL;
    const int t = threadIdx.x;

    float mx = -3.4e38f;
    for (int i = t; i < LL; i += 256) mx = fmaxf(mx, p[i]);
    red[t] = mx; __syncthreads();
    for (int s = 128; s > 0; s >>= 1) {
        if (t < s) red[t] = fmaxf(red[t], red[t + s]);
        __syncthreads();
    }
    mx = red[0]; __syncthreads();

    float sum = 0.f;
    for (int i = t; i < LL; i += 256) sum += __expf(p[i] - mx);
    red[t] = sum; __syncthreads();
    for (int s = 128; s > 0; s >>= 1) {
        if (t < s) red[t] += red[t + s];
        __syncthreads();
    }
    const float inv = 1.0f / red[0];

    unsigned short* q = outbf + row * LL;
    for (int i = t; i < LL; i += 256) q[i] = f32_to_bf16(__expf(p[i] - mx) * inv);
}

// -------- softmax over 32 head channels; write Q TRANSPOSED bf16 -----------
// in : [NB, DD, LL] f32 ; out qt : [NB*HH, LL, 32] bf16 (k contiguous)
__global__ __launch_bounds__(256) void softmax_q_kernel(
    const float* __restrict__ qin, unsigned short* __restrict__ qt) {
    const size_t gid = (size_t)blockIdx.x * 256 + threadIdx.x; // NB*HH*LL total
    const size_t l  = gid % LL;
    const size_t nh = gid / LL;
    const size_t n  = nh / HH, h = nh % HH;

    const float* p = qin + n * DD * LL + (h * 32) * LL + l;
    float vals[32];
    float mx = -3.4e38f;
    #pragma unroll
    for (int k = 0; k < 32; ++k) { vals[k] = p[(size_t)k * LL]; mx = fmaxf(mx, vals[k]); }
    float s = 0.f;
    #pragma unroll
    for (int k = 0; k < 32; ++k) { vals[k] = __expf(vals[k] - mx); s += vals[k]; }
    const float inv = 1.0f / s;
    unsigned short* o = qt + gid * 32;
    #pragma unroll
    for (int k = 0; k < 32; ++k) o[k] = f32_to_bf16(vals[k] * inv);
}

// ------------- context[n,h] = K_h (32xL) @ V_h^T (Lx32) via WMMA -----------
// 1 block per (n,h): 128 thr = 4 waves, each wave one 16x16 output tile.
// K/V tiles (32 rows x 256 cols) staged into LDS by the Tensor Data Mover
// (wave 0 issues two 2D descriptors; TDM pads 16B per row -> pitch 528B).
__global__ __launch_bounds__(128) void context_kernel(
    const unsigned short* __restrict__ kbf, const unsigned short* __restrict__ vbf,
    float* __restrict__ ctx) {
    __shared__ __align__(16) unsigned short Ks[32 * CTX_PITCH];
    __shared__ __align__(16) unsigned short Vs[32 * CTX_PITCH];

    const int nh = blockIdx.x;
    const int n = nh >> 3, h = nh & 7;
    const int wave = threadIdx.x >> 5, lane = threadIdx.x & 31;
    const int it = wave >> 1, jt = wave & 1;       // 2x2 tile grid over 32x32
    const int ml = lane & 15, half = lane >> 4;

    const unsigned short* kbase = kbf + ((size_t)n * DD + h * 32) * LL;
    const unsigned short* vbase = vbf + ((size_t)n * DD + h * 32) * LL;
    const unsigned ks_off = (unsigned)(size_t)(&Ks[0]); // LDS aperture: low 32b = offset
    const unsigned vs_off = (unsigned)(size_t)(&Vs[0]);

    v8f acc = (v8f){0,0,0,0,0,0,0,0};
    for (int l0 = 0; l0 < LL; l0 += 256) {
        __syncthreads();                           // LDS free for overwrite
        if (threadIdx.x < 32) {                    // wave 0 drives the TDM
            tdm_load_2d_bf16(ks_off, kbase + l0, LL, 32, 256, 32, LL, TDM_FLAGS_256COL);
            tdm_load_2d_bf16(vs_off, vbase + l0, LL, 32, 256, 32, LL, TDM_FLAGS_256COL);
            __builtin_amdgcn_s_wait_tensorcnt(0);
        }
        __syncthreads();                           // tiles visible to all waves

        #pragma unroll
        for (int li = 0; li < 256; li += 32) {
            FragBF a, b;
            const unsigned short* kr = &Ks[(it * 16 + ml) * CTX_PITCH];
            a.q[0] = *(const uint4*)(kr + li + 16 * half);
            a.q[1] = *(const uint4*)(kr + li + 16 * half + 8);
            const unsigned short* vr = &Vs[(jt * 16 + ml) * CTX_PITCH];
            b.q[0] = *(const uint4*)(vr + li + 16 * half);
            b.q[1] = *(const uint4*)(vr + li + 16 * half + 8);
            acc = wmma_bf16(a, b, acc);
        }
    }
    float* c = ctx + (size_t)nh * 1024;            // [32 kk][32 vv]
    #pragma unroll
    for (int r = 0; r < 8; ++r)
        c[(it * 16 + r + 8 * half) * 32 + jt * 16 + ml] = acc[r];
}

// ------- att[v,l] = sum_k ctx[k,v] * Q[k,l]  -> aggbf [n, h*32+v, l] -------
__global__ __launch_bounds__(128) void attend_kernel(
    const float* __restrict__ ctx, const unsigned short* __restrict__ qt,
    unsigned short* __restrict__ aggbf) {
    __shared__ unsigned short ctxT[32 * 40];       // ctx^T in bf16, pitch 80B
    const int nh = blockIdx.y;
    const int n = nh >> 3, h = nh & 7;
    const int t = threadIdx.x;

    for (int i = t; i < 1024; i += 128) {          // uniform: 8 iters each
        const int k = i >> 5, v = i & 31;
        ctxT[v * 40 + k] = f32_to_bf16(ctx[(size_t)nh * 1024 + i]);
    }
    __syncthreads();

    const int wave = t >> 5, lane = t & 31;
    const int ml = lane & 15, half = lane >> 4;
    const int l0 = blockIdx.x * 64 + wave * 16;

    FragBF b;                                      // B[k, n=l] from Q^T rows
    const unsigned short* qr = qt + ((size_t)nh * LL + l0 + ml) * 32;
    b.q[0] = *(const uint4*)(qr + 16 * half);
    b.q[1] = *(const uint4*)(qr + 16 * half + 8);

    #pragma unroll
    for (int vt = 0; vt < 2; ++vt) {
        FragBF a;                                  // A[v, k] = ctx^T
        const unsigned short* ar = &ctxT[(vt * 16 + ml) * 40];
        a.q[0] = *(const uint4*)(ar + 16 * half);
        a.q[1] = *(const uint4*)(ar + 16 * half + 8);
        v8f acc = (v8f){0,0,0,0,0,0,0,0};
        acc = wmma_bf16(a, b, acc);
        #pragma unroll
        for (int r = 0; r < 8; ++r) {
            const int v = vt * 16 + r + 8 * half;
            aggbf[(size_t)n * DD * LL + (size_t)(h * 32 + v) * LL + l0 + ml] =
                f32_to_bf16(acc[r]);
        }
    }
}

// ---------------------------------------------------------------------------
extern "C" void kernel_launch(void* const* d_in, const int* in_sizes, int n_in,
                              void* d_out, int out_size, void* d_ws, size_t ws_size,
                              hipStream_t stream) {
    (void)in_sizes; (void)n_in; (void)out_size; (void)ws_size;

    const float* x  = (const float*)d_in[0];
    const float* Wk = (const float*)d_in[1];
    const float* bk = (const float*)d_in[2];
    const float* Wq = (const float*)d_in[3];
    const float* bq = (const float*)d_in[4];
    const float* Wv = (const float*)d_in[5];
    const float* bv = (const float*)d_in[6];
    const float* Wr = (const float*)d_in[7];
    const float* br = (const float*)d_in[8];
    float* out = (float*)d_out;

    const size_t NDL = (size_t)NB * DD * LL;       // 33,554,432 elements
    char* ws = (char*)d_ws;
    size_t off = 0;
    auto carve = [&](size_t bytes) -> char* {
        char* p = ws + off;
        off += (bytes + 255) & ~(size_t)255;
        return p;
    };
    unsigned short* xbf  = (unsigned short*)carve(NDL * 2);          // x in bf16
    unsigned short* wbf  = (unsigned short*)carve(4 * 65536 * 2);    // Wk|Wq|Wv|Wr bf16
    float*          keyF = (float*)carve(NDL * 4);                   // keys f32
    float*          qryF = (float*)carve(NDL * 4);                   // queries f32
    unsigned short* kbf  = (unsigned short*)carve(NDL * 2);          // softmax(K) bf16
    unsigned short* vbf  = (unsigned short*)carve(NDL * 2);          // V bf16
    unsigned short* qtbf = (unsigned short*)carve(NDL * 2);          // softmax(Q)^T bf16
    float*          ctx  = (float*)carve((size_t)NB * HH * 1024 * 4);
    unsigned short* aggbf = (unsigned short*)keyF;                   // reuse dead keys buf

    unsigned short* wkb = wbf + 0 * 65536;
    unsigned short* wqb = wbf + 1 * 65536;
    unsigned short* wvb = wbf + 2 * 65536;
    unsigned short* wrb = wbf + 3 * 65536;

    // 1) convert x and weights to bf16
    convert_bf16_kernel<<<(unsigned)(NDL / 256), 256, 0, stream>>>(x, xbf, (unsigned)NDL);
    convert_bf16_kernel<<<256, 256, 0, stream>>>(Wk, wkb, 65536);
    convert_bf16_kernel<<<256, 256, 0, stream>>>(Wq, wqb, 65536);
    convert_bf16_kernel<<<256, 256, 0, stream>>>(Wv, wvb, 65536);
    convert_bf16_kernel<<<256, 256, 0, stream>>>(Wr, wrb, 65536);

    // 2) projections: keys/queries (f32 for softmax), values (bf16 direct)
    dim3 gg(LL / 512, DD / 16, NB);                // (32, 16, 8)
    gemm_proj_kernel<<<gg, 256, 0, stream>>>(wkb, bk, xbf, keyF, nullptr);
    gemm_proj_kernel<<<gg, 256, 0, stream>>>(wqb, bq, xbf, qryF, nullptr);
    gemm_proj_kernel<<<gg, 256, 0, stream>>>(wvb, bv, xbf, nullptr, vbf);

    // 3) softmaxes
    softmax_rows_kernel<<<NB * DD, 256, 0, stream>>>(keyF, kbf);
    softmax_q_kernel<<<(unsigned)((size_t)NB * HH * LL / 256), 256, 0, stream>>>(qryF, qtbf);

    // 4) context = K @ V^T per (n,h), TDM-staged tiles
    context_kernel<<<NB * HH, 128, 0, stream>>>(kbf, vbf, ctx);

    // 5) att = ctx^T @ Q -> aggregated bf16 feature map
    attend_kernel<<<dim3(LL / 64, NB * HH), 128, 0, stream>>>(ctx, qtbf, aggbf);

    // 6) reprojection into d_out (f32)
    gemm_proj_kernel<<<gg, 256, 0, stream>>>(wrb, br, aggbf, out, nullptr);
}
